// AttLstm_17403207483543
// MI455X (gfx1250) — matrix operand order repaired and checked by reading, not compiled, not run on hardware
//
#include <hip/hip_runtime.h>
#include <hip/hip_bf16.h>

// Problem constants (from reference): B=256, T=2048, I=128, H=128
#define B_  256
#define T_  2048
#define I_  128
#define H_  128
#define G4_ 512           // 4*H gate columns
#define GRS_ 20           // gatesT row stride (floats): 16B-aligned, bank-friendly

typedef __attribute__((ext_vector_type(16))) __bf16 bf16x16;
typedef __attribute__((ext_vector_type(8)))  __bf16 bf16x8;
typedef __attribute__((ext_vector_type(4)))  __bf16 bf16x4;
typedef __attribute__((ext_vector_type(8)))  float  f32x8;
typedef __attribute__((ext_vector_type(4)))  float  f32x4;

__device__ __forceinline__ float fast_rcpf(float x) {
#if __has_builtin(__builtin_amdgcn_rcpf)
    return __builtin_amdgcn_rcpf(x);          // v_rcp_f32, no IEEE div sequence
#else
    return 1.0f / x;
#endif
}
// Branchless activations: 2 TRANS ops (v_exp_f32 + v_rcp_f32), no exec-mask
// divergence (ocml tanhf is branchy and was polluting the hot loop).
__device__ __forceinline__ float fast_sigmoidf(float x) {
    return fast_rcpf(1.0f + __expf(-x));      // x->-inf: exp->inf, rcp->0 ; x->+inf: ->1
}
__device__ __forceinline__ float fast_tanhf(float x) {
    return 1.0f - 2.0f * fast_rcpf(1.0f + __expf(2.0f * x));  // saturates to -1/+1
}

// ---------------------------------------------------------------------------
// Kernel 1: persistent fused LSTM recurrence.
// grid = 16 workgroups (16 batch rows each), block = 512 threads (16 waves).
// Per step: gates[16x512] = [x_t | h_{t-1}] (16x256 bf16) @ [W_ih|W_hh]^T via
// v_wmma_f32_16x16x32_bf16; weights live in VGPRs for all 2048 steps.
// Gates staged through a TRANSPOSED LDS buffer so both the WMMA-side stores
// and the elementwise-side reads are b128 DS ops.
// ---------------------------------------------------------------------------
__launch_bounds__(512, 1)
__global__ void lstm_recurrent_kernel(const float* __restrict__ x,
                                      const float* __restrict__ W_ih,
                                      const float* __restrict__ W_hh,
                                      const float* __restrict__ b_ih,
                                      const float* __restrict__ b_hh,
                                      float* __restrict__ hs,      // [B,T,H]
                                      float* __restrict__ h_last)  // [B,H]
{
    __shared__ __align__(16) __bf16 xh[16][256];       // k<128: x_t ; k>=128: h_{t-1}
    __shared__ __align__(16) float  gatesT[G4_][GRS_]; // [gate col][batch row]
    __shared__ float bias[G4_];

    const int tid  = threadIdx.x;
    const int lane = tid & 31;
    const int wave = tid >> 5;
    const int l16  = lane & 15;
    const int half = (lane < 16) ? 0 : 1;
    const int b0   = blockIdx.x * 16;

    // combined bias (b_ih + b_hh), loaded once
    bias[tid] = b_ih[tid] + b_hh[tid];    // blockDim.x == 512 == G4_

    // ---- Load B-fragments (weights) into registers, kept live across T ----
    // B 32x16 bf16 layout: lane<16 -> col N=l16, k = kbase+0..15 along vector;
    //                      lane>=16 -> col N=l16, k = kbase+16..31.
    bf16x16 bfrag[2][8];
    #pragma unroll
    for (int j = 0; j < 2; ++j) {
        const int col = wave * 32 + j * 16 + l16;           // gate row of W (0..511)
        #pragma unroll
        for (int kc = 0; kc < 8; ++kc) {
            const float* Wrow = (kc < 4) ? (W_ih + (size_t)col * I_)
                                         : (W_hh + (size_t)col * H_);
            const int ks = (kc & 3) * 32 + half * 16;       // 0..112, 16-float run
            const float4* wp = (const float4*)(Wrow + ks);
            float4 w0 = wp[0], w1 = wp[1], w2 = wp[2], w3 = wp[3];
            bf16x16 f;
            f[0]=(__bf16)w0.x;  f[1]=(__bf16)w0.y;  f[2]=(__bf16)w0.z;  f[3]=(__bf16)w0.w;
            f[4]=(__bf16)w1.x;  f[5]=(__bf16)w1.y;  f[6]=(__bf16)w1.z;  f[7]=(__bf16)w1.w;
            f[8]=(__bf16)w2.x;  f[9]=(__bf16)w2.y;  f[10]=(__bf16)w2.z; f[11]=(__bf16)w2.w;
            f[12]=(__bf16)w3.x; f[13]=(__bf16)w3.y; f[14]=(__bf16)w3.z; f[15]=(__bf16)w3.w;
            bfrag[j][kc] = f;
        }
    }

    // ---- elementwise-phase thread mapping: col = tid&127, 4 rows each ----
    const int ecol  = tid & 127;
    const int erow0 = (tid >> 7) * 4;
    float c_state[4] = {0.f, 0.f, 0.f, 0.f};

    // h_0 = 0 in the A operand buffer
    #pragma unroll
    for (int ii = 0; ii < 4; ++ii)
        xh[erow0 + ii][128 + ecol] = (__bf16)0.0f;

    // x-load mapping: 512 threads x float4 = 16 rows x 128 cols
    const int xrow = tid >> 5;
    const int xcol = (tid & 31) * 4;
    const float* xbase = x + ((size_t)(b0 + xrow) * T_) * I_ + xcol;

    for (int t = 0; t < T_; ++t) {
        // stage x_t (fp32 -> bf16) into LDS
        const float4 xv = *(const float4*)(xbase + (size_t)t * I_);
        bf16x4 xb;
        xb[0]=(__bf16)xv.x; xb[1]=(__bf16)xv.y; xb[2]=(__bf16)xv.z; xb[3]=(__bf16)xv.w;
        *(bf16x4*)&xh[xrow][xcol] = xb;
        if (t + 8 < T_)                      // pull future timestep toward L2/L0
            __builtin_prefetch(xbase + (size_t)(t + 8) * I_, 0, 1);

        __syncthreads();                     // x_t and h_{t-1} visible to all waves

        // A-fragments 16x32 bf16: lane<16 -> row l16, K 0..15;
        //                         lane>=16 -> row l16, K 16..31
        bf16x16 afrag[8];
        #pragma unroll
        for (int kc = 0; kc < 8; ++kc) {
            const int ks = kc * 32 + half * 16;
            bf16x8 lo = *(const bf16x8*)&xh[l16][ks];
            bf16x8 hi = *(const bf16x8*)&xh[l16][ks + 8];
            afrag[kc] = __builtin_shufflevector(lo, hi,
                0,1,2,3,4,5,6,7,8,9,10,11,12,13,14,15);
        }

        f32x8 acc0 = {}; f32x8 acc1 = {};
        #pragma unroll
        for (int kc = 0; kc < 8; ++kc) {
            acc0 = __builtin_amdgcn_wmma_f32_16x16x32_bf16(
                false, afrag[kc], false, bfrag[0][kc], (short)0, acc0, false, false);
            acc1 = __builtin_amdgcn_wmma_f32_16x16x32_bf16(
                false, afrag[kc], false, bfrag[1][kc], (short)0, acc1, false, false);
        }

        // C/D layout: VGPR r, lane -> row = r + 8*half, col = l16.
        // Transposed gate staging: rows are contiguous -> 4 x ds_store_b128.
        {
            const int rb = half * 8;
            const int c0 = wave * 32 + l16;
            f32x4 a0lo = __builtin_shufflevector(acc0, acc0, 0, 1, 2, 3);
            f32x4 a0hi = __builtin_shufflevector(acc0, acc0, 4, 5, 6, 7);
            f32x4 a1lo = __builtin_shufflevector(acc1, acc1, 0, 1, 2, 3);
            f32x4 a1hi = __builtin_shufflevector(acc1, acc1, 4, 5, 6, 7);
            *(f32x4*)&gatesT[c0     ][rb]     = a0lo;   // rows rb..rb+3
            *(f32x4*)&gatesT[c0     ][rb + 4] = a0hi;   // rows rb+4..rb+7
            *(f32x4*)&gatesT[c0 + 16][rb]     = a1lo;
            *(f32x4*)&gatesT[c0 + 16][rb + 4] = a1hi;
        }

        __syncthreads();                     // gates complete

        // fused gate nonlinearity + cell/hidden update (c kept in registers);
        // each gate's 4 rows arrive as one ds_load_b128.
        {
            const f32x4 gi4 = *(const f32x4*)&gatesT[      ecol][erow0];
            const f32x4 gf4 = *(const f32x4*)&gatesT[128 + ecol][erow0];
            const f32x4 gg4 = *(const f32x4*)&gatesT[256 + ecol][erow0];
            const f32x4 go4 = *(const f32x4*)&gatesT[384 + ecol][erow0];
            const float bi = bias[ecol],       bf = bias[128 + ecol];
            const float bg = bias[256 + ecol], bo = bias[384 + ecol];
            #pragma unroll
            for (int ii = 0; ii < 4; ++ii) {
                const int r = erow0 + ii;
                const float gi = fast_sigmoidf(gi4[ii] + bi);
                const float gf = fast_sigmoidf(gf4[ii] + bf);
                const float gg = fast_tanhf   (gg4[ii] + bg);
                const float go = fast_sigmoidf(go4[ii] + bo);
                const float c  = gf * c_state[ii] + gi * gg;
                c_state[ii] = c;
                const float h  = go * fast_tanhf(c);
                xh[r][128 + ecol] = (__bf16)h;   // next step's A operand
                hs[((size_t)(b0 + r) * T_ + t) * H_ + ecol] = h;
                if (t == T_ - 1)
                    h_last[(b0 + r) * H_ + ecol] = h;
            }
        }
        // next iteration's first __syncthreads covers xh h-write -> WMMA read
    }
}

// ---------------------------------------------------------------------------
// Kernel 2: attention pooling (raw-score normalization, as in the reference).
// One block per batch element; single pass over hs[b]; 4 timesteps per
// reduction round to amortize barriers.
// ---------------------------------------------------------------------------
__launch_bounds__(128, 1)
__global__ void attn_pool_kernel(const float* __restrict__ hs,
                                 const float* __restrict__ h_last,
                                 float* __restrict__ out)   // [B,1,H]
{
    const int b    = blockIdx.x;
    const int tid  = threadIdx.x;          // = h index (0..127)
    const int lane = tid & 31;
    const int wave = tid >> 5;
    __shared__ __align__(16) float red[4][4];   // [wave][q]

    const float  hl  = h_last[b * H_ + tid];
    const float* hsb = hs + (size_t)b * T_ * H_;
    float ctx = 0.f, den = 0.f;

    for (int t = 0; t < T_; t += 4) {
        float v[4], p[4];
        #pragma unroll
        for (int q = 0; q < 4; ++q) {
            v[q] = hsb[(size_t)(t + q) * H_ + tid];
            p[q] = v[q] * hl;
        }
        #pragma unroll
        for (int off = 16; off > 0; off >>= 1) {
            #pragma unroll
            for (int q = 0; q < 4; ++q)
                p[q] += __shfl_xor(p[q], off, 32);
        }
        if (lane == 0) {
            f32x4 pv = {p[0], p[1], p[2], p[3]};
            *(f32x4*)&red[wave][0] = pv;
        }
        __syncthreads();
        #pragma unroll
        for (int q = 0; q < 4; ++q) {
            const float s = red[0][q] + red[1][q] + red[2][q] + red[3][q];
            ctx += s * v[q];
            den += s;
        }
        __syncthreads();
    }
    out[b * H_ + tid] = ctx / den;
}

// ---------------------------------------------------------------------------
extern "C" void kernel_launch(void* const* d_in, const int* in_sizes, int n_in,
                              void* d_out, int out_size, void* d_ws, size_t ws_size,
                              hipStream_t stream) {
    const float* x    = (const float*)d_in[0];
    const float* W_ih = (const float*)d_in[1];
    const float* W_hh = (const float*)d_in[2];
    const float* b_ih = (const float*)d_in[3];
    const float* b_hh = (const float*)d_in[4];

    float* hs     = (float*)d_ws;                        // B*T*H fp32 = 256 MB
    float* h_last = hs + (size_t)B_ * T_ * H_;           // + B*H fp32

    lstm_recurrent_kernel<<<B_ / 16, 512, 0, stream>>>(x, W_ih, W_hh, b_ih, b_hh,
                                                       hs, h_last);
    attn_pool_kernel<<<B_, 128, 0, stream>>>(hs, h_last, (float*)d_out);
}
